// RegexGRU_57466662421091
// MI455X (gfx1250) — matrix operand order repaired and checked by reading, not compiled, not run on hardware
//
#include <hip/hip_runtime.h>
#include <hip/hip_bf16.h>

// Problem constants (match reference)
#define B_  64
#define S_  512
#define H_  512
#define L_  10

using bf = __hip_bfloat16;

typedef __bf16 bf16x16 __attribute__((ext_vector_type(16)));
typedef __bf16 bf16x8  __attribute__((ext_vector_type(8)));
typedef float  f32x8   __attribute__((ext_vector_type(8)));

static __device__ inline f32x8 splat8(float v) {
    return (f32x8){v, v, v, v, v, v, v, v};
}

static __device__ inline bf16x16 join8(bf16x8 lo, bf16x8 hi) {
    return __builtin_shufflevector(lo, hi, 0,1,2,3,4,5,6,7,8,9,10,11,12,13,14,15);
}

// A-fragment (16x32 bf16, MxK): lane holds row m=lane&15; contiguous
// K range [16*(lane>>4) .. +15]
static __device__ inline bf16x16 loadA(const bf* p) {
    return join8(*(const bf16x8*)p, *(const bf16x8*)(p + 8));
}

// B-fragment (32x16 bf16, KxN): lane holds column n=lane&15; contiguous
// K range [16*(lane>>4) .. +15]
static __device__ inline bf16x16 loadB(const bf* p) {
    return join8(*(const bf16x8*)p, *(const bf16x8*)(p + 8));
}

static __device__ inline f32x8 wmma_bf16(bf16x16 a, bf16x16 b, f32x8 c) {
    // 8 args: (neg_a, A, neg_b, B, c_mod, C, reuse_a, reuse_b)
    return __builtin_amdgcn_wmma_f32_16x16x32_bf16(
        false, a, false, b, (short)0, c, false, false);
}

static __device__ inline float sigmoidf_(float v) {
    return 1.0f / (1.0f + __expf(-v));
}

__global__ void gru_zero_flags(int* ready) {
    if (threadIdx.x < L_) ready[threadIdx.x] = 0;
}

// One workgroup per GRU layer; layers form a software pipeline over timesteps.
// Per step t, wave w (of 16) computes gate columns {c, c+32, c+64} for
// c in {w, w+16}: exactly the r/z/n columns needed to update h[:, 16c..16c+16).
__global__ __launch_bounds__(512, 1)
void gru_pipeline(const int* __restrict__ x, const bf* __restrict__ emb,
                  const bf* __restrict__ w_ih, const bf* __restrict__ w_hh,
                  const bf* __restrict__ b_ih, const bf* __restrict__ b_hh,
                  bf* __restrict__ buf0, bf* __restrict__ buf1,
                  int* __restrict__ ready)
{
    __shared__ bf hbuf[B_ * H_];  // 64 KB: resident hidden state h[64][512]

    const int layer = blockIdx.x;
    const bf* wi = w_ih + (size_t)layer * 3 * H_ * H_;
    const bf* wh = w_hh + (size_t)layer * 3 * H_ * H_;
    const bf* bi = b_ih + layer * 3 * H_;
    const bf* bh = b_hh + layer * 3 * H_;
    bf*       outb = (layer & 1) ? buf1 : buf0;             // out = bufs[l%2]
    const bf* inb  = (layer == 0) ? nullptr
                                  : ((layer & 1) ? buf0 : buf1);  // in = bufs[(l+1)%2]

    const int tid  = threadIdx.x;
    const int w    = tid >> 5;          // wave id 0..15
    const int lane = tid & 31;
    const int l15  = lane & 15;
    const int hi   = lane >> 4;

    // h0 = 0
    for (int i = tid; i < B_ * H_; i += 512) hbuf[i] = __float2bfloat16(0.0f);
    __syncthreads();

    for (int t = 0; t < S_; ++t) {
        if (layer > 0) {
            if (tid == 0) {
                while (__hip_atomic_load(&ready[layer - 1], __ATOMIC_ACQUIRE,
                                         __HIP_MEMORY_SCOPE_AGENT) < t + 1)
                    __builtin_amdgcn_s_sleep(1);
            }
            __syncthreads();
        }

        // Per-lane A-operand row bases (4 M-tiles of 16 batch rows).
        // Layer 0 gathers embedding rows directly (fused embedding lookup).
        const bf* pA[4];
#pragma unroll
        for (int mt = 0; mt < 4; ++mt) {
            const int m = 16 * mt + l15;
            if (layer == 0) {
                const int tok = x[(size_t)m * S_ + t];
                pA[mt] = emb + (size_t)tok * H_;
            } else {
                pA[mt] = inb + ((size_t)m * S_ + t) * H_;
            }
        }
        if (layer > 0 && t + 1 < S_)
            __builtin_prefetch(inb + ((size_t)l15 * S_ + (t + 1)) * H_, 0, 1);

        bf hnew[2][4][8];

#pragma unroll
        for (int p = 0; p < 2; ++p) {
            const int c = w + 16 * p;      // h column tile 0..31
            const int j = 16 * c + l15;    // this lane's h column

            // Gate rows of w_ih / w_hh (weight layout [3H, H], row-major)
            const bf* wiR = wi + (size_t)(j) * H_;
            const bf* wiZ = wi + (size_t)(H_ + j) * H_;
            const bf* wiN = wi + (size_t)(2 * H_ + j) * H_;
            const bf* whR = wh + (size_t)(j) * H_;
            const bf* whZ = wh + (size_t)(H_ + j) * H_;
            const bf* whN = wh + (size_t)(2 * H_ + j) * H_;

            const float br  = __bfloat162float(bi[j]) + __bfloat162float(bh[j]);
            const float bz  = __bfloat162float(bi[H_ + j]) + __bfloat162float(bh[H_ + j]);
            const float bnx = __bfloat162float(bi[2 * H_ + j]);
            const float bnh = __bfloat162float(bh[2 * H_ + j]);

            f32x8 accR[4], accZ[4], accNX[4], accNH[4];
#pragma unroll
            for (int mt = 0; mt < 4; ++mt) {
                accR[mt]  = splat8(br);
                accZ[mt]  = splat8(bz);
                accNX[mt] = splat8(bnx);
                accNH[mt] = splat8(bnh);
            }

            // Phase A: input-side GEMM  x_t @ w_ih^T  (r, z, and xn accumulators)
            for (int kk = 0; kk < H_; kk += 32) {
                const bf16x16 b0 = loadB(wiR + kk + 16 * hi);
                const bf16x16 b1 = loadB(wiZ + kk + 16 * hi);
                const bf16x16 b2 = loadB(wiN + kk + 16 * hi);
#pragma unroll
                for (int mt = 0; mt < 4; ++mt) {
                    const bf16x16 a = loadA(pA[mt] + kk + 16 * hi);
                    accR[mt]  = wmma_bf16(a, b0, accR[mt]);
                    accZ[mt]  = wmma_bf16(a, b1, accZ[mt]);
                    accNX[mt] = wmma_bf16(a, b2, accNX[mt]);
                }
            }
            // Phase B: hidden-side GEMM  h_{t-1} @ w_hh^T  (A from LDS)
            for (int kk = 0; kk < H_; kk += 32) {
                const bf16x16 b0 = loadB(whR + kk + 16 * hi);
                const bf16x16 b1 = loadB(whZ + kk + 16 * hi);
                const bf16x16 b2 = loadB(whN + kk + 16 * hi);
#pragma unroll
                for (int mt = 0; mt < 4; ++mt) {
                    const bf16x16 a = loadA(&hbuf[(16 * mt + l15) * H_ + kk + 16 * hi]);
                    accR[mt]  = wmma_bf16(a, b0, accR[mt]);
                    accZ[mt]  = wmma_bf16(a, b1, accZ[mt]);
                    accNH[mt] = wmma_bf16(a, b2, accNH[mt]);
                }
            }

            // GRU gate fusion. C/D layout: element i of the v8f accumulator is
            // row m = 16*mt + 8*(lane>>4) + i, column j (this lane's column).
#pragma unroll
            for (int mt = 0; mt < 4; ++mt) {
#pragma unroll
                for (int i = 0; i < 8; ++i) {
                    const float r = sigmoidf_(accR[mt][i]);
                    const float z = sigmoidf_(accZ[mt][i]);
                    const float n = tanhf(accNX[mt][i] + r * accNH[mt][i]);
                    const int   m = 16 * mt + 8 * hi + i;
                    const float ho = __bfloat162float(hbuf[m * H_ + j]);
                    hnew[p][mt][i] = __float2bfloat16((1.0f - z) * n + z * ho);
                }
            }
        }

        __syncthreads();  // all reads of h_{t-1} (LDS) complete
#pragma unroll
        for (int p = 0; p < 2; ++p) {
            const int j = 16 * (w + 16 * p) + l15;
#pragma unroll
            for (int mt = 0; mt < 4; ++mt)
#pragma unroll
                for (int i = 0; i < 8; ++i)
                    hbuf[(16 * mt + 8 * hi + i) * H_ + j] = hnew[p][mt][i];
        }
        __syncthreads();  // LDS now holds h_t

        // Cooperative contiguous store of h_t -> seq_out[:, t, :] (128B/thread)
        {
            const int row = tid >> 3;
            const int off = (tid & 7) * 64;
            const uint4* src = (const uint4*)&hbuf[row * H_ + off];
            uint4* dst = (uint4*)(outb + ((size_t)row * S_ + t) * H_ + off);
#pragma unroll
            for (int q = 0; q < 8; ++q) dst[q] = src[q];
        }
        __threadfence();
        __syncthreads();
        if (tid == 0)
            __hip_atomic_store(&ready[layer], t + 1, __ATOMIC_RELEASE,
                               __HIP_MEMORY_SCOPE_AGENT);
    }
}

// EOS gather + fc dot + sigmoid epilogue (tiny)
__global__ void gru_finalize(const int* __restrict__ x, const bf* __restrict__ seq,
                             const bf* __restrict__ fcw, const bf* __restrict__ fcb,
                             bf* __restrict__ outp)
{
    const int b = threadIdx.x;
    if (b >= B_) return;
    int eos = 0;
    for (int s = 0; s < S_; ++s) {
        if (x[(size_t)b * S_ + s] == 2) { eos = s; break; }
    }
    float acc = __bfloat162float(fcb[0]);
    const bf* hrow = seq + ((size_t)b * S_ + eos) * H_;
    for (int k = 0; k < H_; ++k)
        acc += __bfloat162float(hrow[k]) * __bfloat162float(fcw[k]);
    outp[b] = __float2bfloat16(1.0f / (1.0f + __expf(-acc)));
}

extern "C" void kernel_launch(void* const* d_in, const int* in_sizes, int n_in,
                              void* d_out, int out_size, void* d_ws, size_t ws_size,
                              hipStream_t stream)
{
    (void)in_sizes; (void)n_in; (void)out_size; (void)ws_size;
    const int* x  = (const int*)d_in[0];
    const bf* emb = (const bf*)d_in[1];
    const bf* wih = (const bf*)d_in[2];
    const bf* whh = (const bf*)d_in[3];
    const bf* bih = (const bf*)d_in[4];
    const bf* bhh = (const bf*)d_in[5];
    const bf* fcw = (const bf*)d_in[6];
    const bf* fcb = (const bf*)d_in[7];
    bf* outp = (bf*)d_out;

    char* ws = (char*)d_ws;
    bf* buf0 = (bf*)ws;                            // [B,S,H] bf16 (32 MB)
    bf* buf1 = buf0 + (size_t)B_ * S_ * H_;        // [B,S,H] bf16 (32 MB)
    int* ready = (int*)(buf1 + (size_t)B_ * S_ * H_);  // L ready-flags

    gru_zero_flags<<<1, 32, 0, stream>>>(ready);
    gru_pipeline<<<L_, 512, 0, stream>>>(x, emb, wih, whh, bih, bhh,
                                         buf0, buf1, ready);
    // layer 9 (odd) wrote buf1
    gru_finalize<<<1, B_, 0, stream>>>(x, buf1, fcw, fcb, outp);
}
